// NonLocalBlockND_498216206348
// MI455X (gfx1250) — matrix operand order repaired and checked by reading, not compiled, not run on hardware
//
#include <hip/hip_runtime.h>
#include <hip/hip_bf16.h>
#include <stdint.h>

typedef __bf16 bf16_t;
typedef __attribute__((ext_vector_type(16))) __bf16 v16bf;
typedef __attribute__((ext_vector_type(8)))  float  v8f;

union V16 { v16bf v; uint32_t u[8]; bf16_t h[16]; };

#define B_   16
#define C_   256
#define CI_  128
#define N_   4096   // H*W
#define M_   1024   // pooled spatial

// ---------------------------------------------------------------------------
// 0) Pack weights to bf16: wAll = [theta_w; phi_w; g_w] as [384,256]; Ww [256,128]
// ---------------------------------------------------------------------------
__global__ void k_prep(const float* tw, const float* pw, const float* gw,
                       const float* Ww, bf16_t* wAllB, bf16_t* WwB) {
  int idx = blockIdx.x * 256 + threadIdx.x;
  const int nAll = 384 * 256;
  if (idx < nAll) {
    int r = idx >> 8, c = idx & 255;
    float v = (r < 128) ? tw[r * 256 + c]
            : (r < 256) ? pw[(r - 128) * 256 + c]
                        : gw[(r - 256) * 256 + c];
    wAllB[idx] = (bf16_t)v;
  } else {
    int j = idx - nAll;
    if (j < C_ * CI_) WwB[j] = (bf16_t)Ww[j];
  }
}

// ---------------------------------------------------------------------------
// 1) Projections: per batch, [384,256]bf16 @ x[b][256,4096]f32->bf16.
//    8 waves/block; block tile 128 rows x 64 cols. x tile staged once per
//    k-step in LDS as bf16 (pair-packed ds_store_b32), reused by all 8 waves.
//    theta rows -> thetaT [B][N][CI] (transposed, +bias)
//    phi   rows -> phifull [B][CI][N] (+bias);  g rows -> gfull likewise.
// ---------------------------------------------------------------------------
__global__ __launch_bounds__(256) void k_proj(
    const float* __restrict__ x, const bf16_t* __restrict__ wAllB,
    const float* __restrict__ tb, const float* __restrict__ pb,
    const float* __restrict__ gb,
    bf16_t* __restrict__ thetaT, bf16_t* __restrict__ phifull,
    bf16_t* __restrict__ gfull) {
  // [n][c] tile, padded to 36 bf16 (72B row stride): 4-byte aligned rows,
  // bank-conflict-free staging stores (18 u32-banks stride).
  __shared__ __align__(16) bf16_t xs[64][36];
  const int b     = blockIdx.z;
  const int nbase = blockIdx.x * 64;
  const int t     = threadIdx.x;
  const int wave  = t >> 5;
  const int lane  = t & 31;
  const int half  = lane >> 4, l15 = lane & 15;
  const int mbase = blockIdx.y * 128 + wave * 16;

  v8f acc[4] = {};
  const uint32_t* wrow = reinterpret_cast<const uint32_t*>(wAllB + (mbase + l15) * 256);
  const float* xb = x + (size_t)b * C_ * N_;

  for (int k = 0; k < 256; k += 32) {
    __syncthreads();   // previous-iteration readers done
#pragma unroll
    for (int j = 0; j < 4; ++j) {
      int p  = t + 256 * j;        // 0..1023 pair index over 32(K) x 64(N) tile
      int cl = (p >> 6) * 2;       // even 0..30
      int nl = p & 63;
      float lo = xb[(size_t)(k + cl) * N_ + nbase + nl];
      float hi = xb[(size_t)(k + cl + 1) * N_ + nbase + nl];
      union { uint32_t u; bf16_t h[2]; } pk;
      pk.h[0] = (bf16_t)lo;
      pk.h[1] = (bf16_t)hi;
      *reinterpret_cast<uint32_t*>(&xs[nl][cl]) = pk.u;   // one ds_store_b32
    }
    __syncthreads();

    V16 a;
#pragma unroll
    for (int j = 0; j < 8; ++j) {
      int kp = 2 * j + half * 16 + k; // A-frag K-pair map
      a.u[j] = wrow[kp >> 1];
    }
    V16 bb[4];                       // hoist all B-frag LDS loads ahead of WMMAs
#pragma unroll
    for (int f = 0; f < 4; ++f) {
      const uint32_t* ls =
          reinterpret_cast<const uint32_t*>(&xs[f * 16 + l15][half * 16]);
#pragma unroll
      for (int j = 0; j < 8; ++j) bb[f].u[j] = ls[j];
    }
#pragma unroll
    for (int f = 0; f < 4; ++f)
      acc[f] = __builtin_amdgcn_wmma_f32_16x16x32_bf16(
          false, a.v, false, bb[f].v, (short)0, acc[f], false, false);
  }
#pragma unroll
  for (int f = 0; f < 4; ++f) {
    int n = nbase + f * 16 + l15;
#pragma unroll
    for (int r = 0; r < 8; ++r) {
      int ch = mbase + r + half * 8;                          // D row map
      float v = acc[f][r];
      if (ch < 128) {
        v += tb[ch];
        thetaT[((size_t)b * N_ + n) * CI_ + ch] = (bf16_t)v;
      } else if (ch < 256) {
        int c2 = ch - 128; v += pb[c2];
        phifull[((size_t)b * CI_ + c2) * N_ + n] = (bf16_t)v;
      } else {
        int c2 = ch - 256; v += gb[c2];
        gfull[((size_t)b * CI_ + c2) * N_ + n] = (bf16_t)v;
      }
    }
  }
}

// ---------------------------------------------------------------------------
// 2) 2x2 maxpool: phiB [B][M][CI] (m-major, for QK B-frags),
//                 gB   [B][CI][M] (c-major, for PV B-frags)
// ---------------------------------------------------------------------------
__global__ void k_pool(const bf16_t* __restrict__ phifull,
                       const bf16_t* __restrict__ gfull,
                       bf16_t* __restrict__ phiB, bf16_t* __restrict__ gB) {
  int idx = blockIdx.x * 256 + threadIdx.x;        // B*CI*M = 2^21
  if (idx >= B_ * CI_ * M_) return;
  int m = idx & (M_ - 1);
  int c = (idx >> 10) & (CI_ - 1);
  int b = idx >> 17;
  int ph = m >> 5, pw = m & 31;
  size_t base = ((size_t)b * CI_ + c) * N_ + (size_t)(2 * ph) * 64 + 2 * pw;
  float p0 = (float)phifull[base],      p1 = (float)phifull[base + 1];
  float p2 = (float)phifull[base + 64], p3 = (float)phifull[base + 65];
  phiB[((size_t)b * M_ + m) * CI_ + c] = (bf16_t)fmaxf(fmaxf(p0, p1), fmaxf(p2, p3));
  float g0 = (float)gfull[base],      g1 = (float)gfull[base + 1];
  float g2 = (float)gfull[base + 64], g3 = (float)gfull[base + 65];
  gB[((size_t)b * CI_ + c) * M_ + m] = (bf16_t)fmaxf(fmaxf(g0, g1), fmaxf(g2, g3));
}

// ---------------------------------------------------------------------------
// 3) Fused flash attention: one wave per 16 queries, stream M in tiles of 32,
//    online softmax (never materializes the 4096x1024 attn matrix).
// ---------------------------------------------------------------------------
__global__ __launch_bounds__(128) void k_attn(
    const bf16_t* __restrict__ thetaT, const bf16_t* __restrict__ phiB,
    const bf16_t* __restrict__ gB, bf16_t* __restrict__ yB) {
  __shared__ __align__(16) bf16_t pbuf[4][16][32];  // per-wave P staging (D->A relayout)
  const int wave = threadIdx.x >> 5;
  const int lane = threadIdx.x & 31;
  const int half = lane >> 4, l15 = lane & 15;
  const int wg    = blockIdx.x * 4 + wave;          // 0..4095
  const int b     = wg >> 8;
  const int qbase = (wg & 255) * 16;

  V16 aq[4];                                        // Q: 16 rows x 128 channels
  {
    const uint32_t* qrow =
        reinterpret_cast<const uint32_t*>(thetaT + ((size_t)b * N_ + qbase + l15) * CI_);
#pragma unroll
    for (int k = 0; k < 4; ++k)
#pragma unroll
      for (int j = 0; j < 8; ++j) {
        int kp = 2 * j + half * 16 + k * 32;
        aq[k].u[j] = qrow[kp >> 1];
      }
  }
  v8f yacc[8] = {};                                 // 16 rows x 128 out channels
  float mrow[8], lrow[8];
#pragma unroll
  for (int r = 0; r < 8; ++r) { mrow[r] = -1e30f; lrow[r] = 0.f; }

  const bf16_t* phib = phiB + (size_t)b * M_ * CI_;
  const bf16_t* gb   = gB + (size_t)b * CI_ * M_;
  const uint32_t* phi32 = reinterpret_cast<const uint32_t*>(phib);
  const uint32_t* g32   = reinterpret_cast<const uint32_t*>(gb);

  for (int m0 = 0; m0 < M_; m0 += 32) {
    // prefetch next tile of phi / g while this tile's WMMAs run
    if (m0 + 32 < M_) {
      __builtin_prefetch(phib + (m0 + 32 + (lane & 31)) * CI_, 0, 0);
      __builtin_prefetch(gb + (size_t)(lane * 4) * M_ + m0 + 32, 0, 0);
    }
    v8f s[2] = {};
#pragma unroll
    for (int tt = 0; tt < 2; ++tt) {
      int mcol = m0 + tt * 16 + l15;
#pragma unroll
      for (int k = 0; k < 4; ++k) {
        V16 bp;
        const uint32_t* pp = phi32 + ((mcol * CI_ + k * 32 + half * 16) >> 1);
#pragma unroll
        for (int j = 0; j < 8; ++j) bp.u[j] = pp[j];
        s[tt] = __builtin_amdgcn_wmma_f32_16x16x32_bf16(
            false, aq[k].v, false, bp.v, (short)0, s[tt], false, false);
      }
    }
    // online softmax per row (rows live in 16-lane halves of the D layout)
#pragma unroll
    for (int r = 0; r < 8; ++r) {
      float v0 = s[0][r], v1 = s[1][r];
      float mx = fmaxf(v0, v1);
#pragma unroll
      for (int d = 8; d >= 1; d >>= 1) mx = fmaxf(mx, __shfl_xor(mx, d, 16));
      float mnew  = fmaxf(mrow[r], mx);
      float scale = __expf(mrow[r] - mnew);
      float p0 = __expf(v0 - mnew);
      float p1 = __expf(v1 - mnew);
      float ps = p0 + p1;
#pragma unroll
      for (int d = 8; d >= 1; d >>= 1) ps += __shfl_xor(ps, d, 16);
      lrow[r] = lrow[r] * scale + ps;
      mrow[r] = mnew;
#pragma unroll
      for (int f = 0; f < 8; ++f) yacc[f][r] = yacc[f][r] * scale;
      int row = r + half * 8;
      pbuf[wave][row][l15]      = (bf16_t)p0;
      pbuf[wave][row][16 + l15] = (bf16_t)p1;
    }
    V16 ap;                                         // P as A-frag (16x32, K=m)
    {
      const uint32_t* pl = reinterpret_cast<const uint32_t*>(&pbuf[wave][l15][0]);
#pragma unroll
      for (int j = 0; j < 8; ++j) {
        int kp = 2 * j + half * 16;
        ap.u[j] = pl[kp >> 1];
      }
    }
#pragma unroll
    for (int f = 0; f < 8; ++f) {
      V16 bg;
      int ccol = f * 16 + l15;
      const uint32_t* gg = g32 + ((ccol * M_ + m0 + half * 16) >> 1);
#pragma unroll
      for (int j = 0; j < 8; ++j) bg.u[j] = gg[j];
      yacc[f] = __builtin_amdgcn_wmma_f32_16x16x32_bf16(
          false, ap.v, false, bg.v, (short)0, yacc[f], false, false);
    }
  }
#pragma unroll
  for (int r = 0; r < 8; ++r) {
    int n = qbase + r + half * 8;
    float inv = 1.0f / lrow[r];
#pragma unroll
    for (int f = 0; f < 8; ++f) {
      int c = f * 16 + l15;
      yB[((size_t)b * N_ + n) * CI_ + c] = (bf16_t)(yacc[f][r] * inv);
    }
  }
}

// ---------------------------------------------------------------------------
// 4) Trailing 1x1 conv: Ww[256,128]bf16 @ y[b][4096,128]bf16 -> wy f32 (+bias)
// ---------------------------------------------------------------------------
__global__ __launch_bounds__(128) void k_conv2(
    const bf16_t* __restrict__ WwB, const bf16_t* __restrict__ yB,
    const float* __restrict__ Wb, float* __restrict__ wy) {
  const int b     = blockIdx.z;
  const int nbase = blockIdx.x * 64;
  const int wave  = threadIdx.x >> 5;
  const int lane  = threadIdx.x & 31;
  const int half  = lane >> 4, l15 = lane & 15;
  const int mbase = blockIdx.y * 64 + wave * 16;

  v8f acc[4] = {};
  const uint32_t* wrow = reinterpret_cast<const uint32_t*>(WwB + (mbase + l15) * CI_);
  const uint32_t* y32  = reinterpret_cast<const uint32_t*>(yB + (size_t)b * N_ * CI_);

  for (int k = 0; k < CI_; k += 32) {
    V16 a;
#pragma unroll
    for (int j = 0; j < 8; ++j) {
      int kp = 2 * j + half * 16 + k;
      a.u[j] = wrow[kp >> 1];
    }
#pragma unroll
    for (int f = 0; f < 4; ++f) {
      V16 bb;
      int n = nbase + f * 16 + l15;
      const uint32_t* pp = y32 + ((n * CI_ + k + half * 16) >> 1);
#pragma unroll
      for (int j = 0; j < 8; ++j) bb.u[j] = pp[j];
      acc[f] = __builtin_amdgcn_wmma_f32_16x16x32_bf16(
          false, a.v, false, bb.v, (short)0, acc[f], false, false);
    }
  }
#pragma unroll
  for (int f = 0; f < 4; ++f) {
    int n = nbase + f * 16 + l15;
#pragma unroll
    for (int r = 0; r < 8; ++r) {
      int ch = mbase + r + half * 8;
      wy[((size_t)b * C_ + ch) * N_ + n] = acc[f][r] + Wb[ch];
    }
  }
}

// ---------------------------------------------------------------------------
// 5) BatchNorm statistics: one block per channel
// ---------------------------------------------------------------------------
__global__ void k_bnstats(const float* __restrict__ wy, float* __restrict__ stats) {
  __shared__ float s1[256], s2[256];
  int ch = blockIdx.x, t = threadIdx.x;
  float sum = 0.f, sq = 0.f;
  for (int i = t; i < B_ * N_; i += 256) {
    int b = i >> 12, n = i & (N_ - 1);
    float v = wy[((size_t)b * C_ + ch) * N_ + n];
    sum += v; sq += v * v;
  }
  s1[t] = sum; s2[t] = sq;
  __syncthreads();
  for (int d = 128; d; d >>= 1) {
    if (t < d) { s1[t] += s1[t + d]; s2[t] += s2[t + d]; }
    __syncthreads();
  }
  if (t == 0) {
    const float invN = 1.0f / (float)(B_ * N_);
    float mean = s1[0] * invN;
    float var  = s2[0] * invN - mean * mean;
    stats[ch]       = mean;
    stats[256 + ch] = rsqrtf(var + 1e-5f);
  }
}

// ---------------------------------------------------------------------------
// 6) Normalize + affine + residual
// ---------------------------------------------------------------------------
__global__ void k_bnapply(const float* __restrict__ wy, const float* __restrict__ x,
                          const float* __restrict__ stats,
                          const float* __restrict__ gamma,
                          const float* __restrict__ beta, float* __restrict__ out) {
  size_t idx = (size_t)blockIdx.x * 256 + threadIdx.x;
  if (idx >= (size_t)B_ * C_ * N_) return;
  int ch = (int)((idx >> 12) & (C_ - 1));
  float v = (wy[idx] - stats[ch]) * stats[256 + ch] * gamma[ch] + beta[ch];
  out[idx] = v + x[idx];
}

// ---------------------------------------------------------------------------
extern "C" void kernel_launch(void* const* d_in, const int* in_sizes, int n_in,
                              void* d_out, int out_size, void* d_ws, size_t ws_size,
                              hipStream_t stream) {
  (void)in_sizes; (void)n_in; (void)out_size; (void)ws_size;
  const float* x        = (const float*)d_in[0];
  const float* theta_w  = (const float*)d_in[1];
  const float* theta_b  = (const float*)d_in[2];
  const float* phi_w    = (const float*)d_in[3];
  const float* phi_b    = (const float*)d_in[4];
  const float* g_w      = (const float*)d_in[5];
  const float* g_b      = (const float*)d_in[6];
  const float* W_w      = (const float*)d_in[7];
  const float* W_b      = (const float*)d_in[8];
  const float* bn_gamma = (const float*)d_in[9];
  const float* bn_beta  = (const float*)d_in[10];

  char* ws = (char*)d_ws;
  bf16_t* wAllB   = (bf16_t*)(ws + 0);          //  384*256*2 = 196608
  bf16_t* WwB     = (bf16_t*)(ws + 196608);     //  256*128*2 =  65536
  float*  stats   = (float*)(ws + 262144);      //  512 f32
  bf16_t* thetaT  = (bf16_t*)(ws + 1048576);    //  16 MiB  [B][N][CI]
  bf16_t* phiB    = (bf16_t*)(ws + 17825792);   //   4 MiB  [B][M][CI]
  bf16_t* gB      = (bf16_t*)(ws + 22020096);   //   4 MiB  [B][CI][M]
  bf16_t* yB      = (bf16_t*)(ws + 26214400);   //  16 MiB  [B][N][CI]
  bf16_t* phifull = (bf16_t*)(ws + 42991616);   //  16 MiB  [B][CI][N]
  bf16_t* gfull   = (bf16_t*)(ws + 59768832);   //  16 MiB  [B][CI][N]
  float*  wy      = (float*)(ws + 42991616);    //  64 MiB, overlaps dead phifull/gfull

  k_prep<<<512, 256, 0, stream>>>(theta_w, phi_w, g_w, W_w, wAllB, WwB);

  dim3 gproj(64, 3, 16);
  k_proj<<<gproj, 256, 0, stream>>>(x, wAllB, theta_b, phi_b, g_b,
                                    thetaT, phifull, gfull);

  k_pool<<<8192, 256, 0, stream>>>(phifull, gfull, phiB, gB);

  k_attn<<<1024, 128, 0, stream>>>(thetaT, phiB, gB, yB);

  dim3 gconv(64, 4, 16);
  k_conv2<<<gconv, 128, 0, stream>>>(WwB, yB, W_b, wy);

  k_bnstats<<<256, 256, 0, stream>>>(wy, stats);

  k_bnapply<<<65536, 256, 0, stream>>>(wy, x, stats, bn_gamma, bn_beta,
                                       (float*)d_out);
}
